// LinearAttention_22711787061342
// MI455X (gfx1250) — matrix operand order repaired and checked by reading, not compiled, not run on hardware
//
#include <hip/hip_runtime.h>
#include <hip/hip_bf16.h>
#include <math.h>
#include <stdint.h>

// ---------------------------------------------------------------------------
// LinearAttention on gfx1250 (CDNA5, wave32, WMMA bf16 -> f32 accum)
//
// Shapes: B=8, H=W=128, C=128, heads=4, head_dim=32, N=16384, inner=128.
// Scrambled-reshape insight: row (h,d) of the [B,4,32,N] view is the
// contiguous tile qkv[b, (h*32+d)*128 : (h*32+d)*128+128, 0:128].
//
// Pipeline:
//  K1  k_qkv  : per (b, tile r) 128x384 = (128x128)@(128x384) WMMA.
//               q -> bf16 raw, v -> bf16, k -> fused in-WG softmax.
//  K2  k_ctx  : per (b,h) ctxT[e][d] = sum_n ksoft[d,n]*v[e,n], K=16384 WMMA,
//               fed by double-buffered GLOBAL_LOAD_ASYNC_TO_LDS_B128.
//  K3  k_qsoft: softmax over d (stride 128*128 elems) in-place + *1/sqrt(32).
//  K4  k_out  : per (b,h,s_local): attn = ctxT(32x32)@Q(32x128) WMMA,
//               out = attn@w_out WMMA, +bias, LayerNorm(eps=1e-3), +x.
// ---------------------------------------------------------------------------

typedef __attribute__((ext_vector_type(16))) __bf16 v16bf;
typedef __attribute__((ext_vector_type(8)))  float  v8f;
typedef int v4i_ __attribute__((vector_size(16)));   // matches builtin's V4i

#define CH 128
#define NSP 16384          // spatial positions per batch image
#define INV_SCALE 0.17677669529663687f  // 32^-0.5
#define LN_EPS 1e-3f

#if defined(__gfx1250__) && __has_builtin(__builtin_amdgcn_global_load_async_to_lds_b128) && __has_builtin(__builtin_amdgcn_s_wait_asynccnt)
#define USE_ASYNC 1
#else
#define USE_ASYNC 0
#endif

#define ASG __attribute__((address_space(1)))
#define ASL __attribute__((address_space(3)))

#if USE_ASYNC
// 16-byte async global->LDS copy. AS1 pointer == generic bits for global
// memory; low 32 bits of a generic LDS pointer are the DS offset (aperture
// scheme, CDNA5 ISA). Builtin wants V4i pointees.
__device__ __forceinline__ void async_cp16(const __bf16* g, __bf16* l) {
  __builtin_amdgcn_global_load_async_to_lds_b128(
      (ASG v4i_*)(uintptr_t)g, (ASL v4i_*)(uint32_t)(uintptr_t)l, 0, 0);
}
#endif

// ---- WMMA fragment builders (CDNA5 16-bit layouts, wave32) -----------------
// A 16x32 (MxK): lane m=L%16; kb = (L<16)?0:8; elem j<8 -> K=kb+j,
//                elem 8+j -> K=kb+16+j.   (row must be contiguous in K)
__device__ __forceinline__ v16bf frag_a(const __bf16* __restrict__ row, int k0) {
  const int lane = threadIdx.x & 31;
  const int kb   = (lane & 16) ? 8 : 0;
  v16bf r;
#pragma unroll
  for (int j = 0; j < 8; ++j) {
    r[j]     = row[k0 + kb + j];
    r[8 + j] = row[k0 + kb + 16 + j];
  }
  return r;
}

// B 32x16 (KxN): lane n=L%16; kb = (L<16)?0:8; elem j<8 -> K=kb+j,
//                elem 8+j -> K=kb+16+j.
// `coln` points to 32+ contiguous K values for column n (transposed storage).
__device__ __forceinline__ v16bf frag_b(const __bf16* __restrict__ coln, int k0) {
  const int lane = threadIdx.x & 31;
  const int kb   = (lane & 16) ? 8 : 0;
  v16bf r;
#pragma unroll
  for (int j = 0; j < 8; ++j) {
    r[j]     = coln[k0 + kb + j];
    r[8 + j] = coln[k0 + kb + 16 + j];
  }
  return r;
}

__device__ __forceinline__ v8f wmma_bf16(v16bf a, v16bf b, v8f c) {
  return __builtin_amdgcn_wmma_f32_16x16x32_bf16(false, a, false, b,
                                                 (short)0, c, false, false);
}

// D/C f32 16x16: elem v -> m = v + (L<16?0:8), n = L%16.
__device__ __forceinline__ void d_coords(int v, int& m, int& n) {
  const int lane = threadIdx.x & 31;
  m = v + ((lane & 16) ? 8 : 0);
  n = lane & 15;
}

// ===========================================================================
// K1: QKV projection + fused k-softmax.  grid: B*128 WGs x 256 threads.
// ===========================================================================
__global__ void __launch_bounds__(256)
k_qkv(const float* __restrict__ x, const float* __restrict__ w_qkv,
      __bf16* __restrict__ qb, __bf16* __restrict__ ks, __bf16* __restrict__ vb) {
  __shared__ __bf16 xs[128][CH];     // x tile,   row = s_local, col = c (32KB)
  __shared__ __bf16 wt[128][CH];     // wt[f][c]  transposed weight part (32KB)
  __shared__ __bf16 pbuf[128][CH];   // part output staging (32KB)
  __shared__ float  redv[8];
  __shared__ float  stat[2];

  const int tid  = threadIdx.x;
  const int lane = tid & 31;
  const int wave = tid >> 5;
  const int b    = blockIdx.x >> 7;
  const int r    = blockIdx.x & 127;          // tile = (head*32 + d)
  const size_t row0 = (size_t)b * NSP + (size_t)r * 128;

  // stage x tile (f32 -> bf16)
#pragma unroll 4
  for (int t = 0; t < 64; ++t) {
    int idx = tid + 256 * t;                  // idx = sl*128 + c
    int sl = idx >> 7, c = idx & 127;
    xs[sl][c] = (__bf16)x[(row0 + sl) * CH + c];
  }

  for (int p = 0; p < 3; ++p) {               // p: 0=q 1=k 2=v
    // stage transposed weight part: wt[f][c] = w_qkv[c][p*128+f]
#pragma unroll 4
    for (int t = 0; t < 64; ++t) {
      int idx = tid + 256 * t;                // idx = c*128 + f (coalesced reads)
      int c = idx >> 7, f = idx & 127;
      wt[f][c] = (__bf16)w_qkv[(size_t)c * 384 + p * 128 + f];
    }
    __syncthreads();

    // GEMM: wave w owns m_tile = w, all 8 n_tiles. K = 128.
    v8f acc[8];
#pragma unroll
    for (int nt = 0; nt < 8; ++nt) acc[nt] = (v8f)0.f;
    const __bf16* arow = &xs[wave * 16 + (lane & 15)][0];
#pragma unroll
    for (int k0 = 0; k0 < 128; k0 += 32) {
      v16bf a = frag_a(arow, k0);
#pragma unroll
      for (int nt = 0; nt < 8; ++nt) {
        v16bf bf = frag_b(&wt[nt * 16 + (lane & 15)][0], k0);
        acc[nt] = wmma_bf16(a, bf, acc[nt]);
      }
    }
    // stage result tile (bf16)
#pragma unroll
    for (int nt = 0; nt < 8; ++nt) {
#pragma unroll
      for (int v = 0; v < 8; ++v) {
        int m, n; d_coords(v, m, n);
        pbuf[wave * 16 + m][nt * 16 + n] = (__bf16)acc[nt][v];
      }
    }
    __syncthreads();

    if (p == 1) {
      // k part: softmax over all 16384 elements of this tile (axis=-1).
      float mx = -INFINITY;
#pragma unroll 4
      for (int t = 0; t < 64; ++t) {
        int idx = tid + 256 * t;
        mx = fmaxf(mx, (float)pbuf[idx >> 7][idx & 127]);
      }
#pragma unroll
      for (int o = 16; o > 0; o >>= 1) mx = fmaxf(mx, __shfl_xor(mx, o, 32));
      if (lane == 0) redv[wave] = mx;
      __syncthreads();
      if (tid == 0) {
        float m2 = redv[0];
#pragma unroll
        for (int i = 1; i < 8; ++i) m2 = fmaxf(m2, redv[i]);
        stat[0] = m2;
      }
      __syncthreads();
      const float gmax = stat[0];

      float sum = 0.f;
#pragma unroll 4
      for (int t = 0; t < 64; ++t) {
        int idx = tid + 256 * t;
        sum += expf((float)pbuf[idx >> 7][idx & 127] - gmax);
      }
#pragma unroll
      for (int o = 16; o > 0; o >>= 1) sum += __shfl_xor(sum, o, 32);
      if (lane == 0) redv[wave] = sum;
      __syncthreads();
      if (tid == 0) {
        float s2 = 0.f;
#pragma unroll
        for (int i = 0; i < 8; ++i) s2 += redv[i];
        stat[1] = 1.f / s2;
      }
      __syncthreads();
      const float rsum = stat[1];
#pragma unroll 4
      for (int t = 0; t < 64; ++t) {
        int idx = tid + 256 * t;
        int sl = idx >> 7, c = idx & 127;
        float v = expf((float)pbuf[sl][c] - gmax) * rsum;
        ks[(row0 + sl) * CH + c] = (__bf16)v;
      }
    } else {
      __bf16* dst = (p == 0) ? qb : vb;
#pragma unroll 4
      for (int t = 0; t < 64; ++t) {
        int idx = tid + 256 * t;
        int sl = idx >> 7, c = idx & 127;
        dst[(row0 + sl) * CH + c] = pbuf[sl][c];
      }
    }
    __syncthreads();  // protect wt/pbuf reuse for next part
  }
}

// ===========================================================================
// K2: context ctxT[b,h,e,d] = sum_n ksoft[d,n] * v[e,n].  K = 16384.
// grid: 32 WGs (b*4+h) x 128 threads (4 waves, one 16x16 tile each).
// Async path: double-buffered GLOBAL_LOAD_ASYNC_TO_LDS_B128 staging of
// 32x512 bf16 chunks of both operands, overlapped with WMMA consumption.
// ===========================================================================
#define CK 512                       // K-chunk (n values per stage)
#define NCHUNK (NSP / CK)            // 32 chunks
#define UNITS (32 * CK * 2 / 16)     // 16B units per matrix chunk = 2048

__global__ void __launch_bounds__(128)
k_ctx(const __bf16* __restrict__ ks, const __bf16* __restrict__ vb,
      float* __restrict__ ctxT) {
  const int tid  = threadIdx.x;
  const int lane = tid & 31;
  const int wave = tid >> 5;
  const int b = blockIdx.x >> 2;
  const int h = blockIdx.x & 3;
  const int mt = wave >> 1;                   // d half
  const int nt = wave & 1;                    // e half

  // element index of row 0 of this head's 32-row group (rows are 16384 elems)
  const size_t aBase = ((size_t)b * NSP + (size_t)(h * 32) * 128) * CH;

#if USE_ASYNC
  __shared__ __bf16 abuf[2][32][CK];          // 2 x 32KB
  __shared__ __bf16 bbuf[2][32][CK];          // 2 x 32KB

  // issue one chunk's async copies (all 128 threads, uniform trip count)
  auto issue = [&](int c, int bf) {
#pragma unroll 4
    for (int u = tid; u < UNITS; u += 128) {
      int d     = u >> 6;                     // 64 x 16B units per row
      int elem  = (u & 63) * 8;               // 8 bf16 per 16B
      size_t ge = aBase + (size_t)d * NSP + (size_t)c * CK + elem;
      async_cp16(ks + ge, &abuf[bf][d][elem]);
      async_cp16(vb + ge, &bbuf[bf][d][elem]);
    }
  };

  v8f acc = (v8f)0.f;
  issue(0, 0);
  for (int c = 0; c < NCHUNK; ++c) {
    const int bf = c & 1;
    __builtin_amdgcn_s_wait_asynccnt(0);      // my chunk-c copies landed
    __syncthreads();                          // everyone's landed; prev buf free
    if (c + 1 < NCHUNK) issue(c + 1, bf ^ 1);
    const __bf16* arow = &abuf[bf][mt * 16 + (lane & 15)][0];
    const __bf16* brow = &bbuf[bf][nt * 16 + (lane & 15)][0];
#pragma unroll
    for (int kl = 0; kl < CK; kl += 32) {
      v16bf a  = frag_a(arow, kl);
      v16bf bm = frag_b(brow, kl);
      acc = wmma_bf16(a, bm, acc);
    }
  }
#else
  // fallback: per-lane fragment loads straight from global (L2-resident)
  const int d0 = h * 32 + mt * 16 + (lane & 15);
  const int e0 = h * 32 + nt * 16 + (lane & 15);
  const __bf16* arow = ks + ((size_t)b * NSP + (size_t)d0 * 128) * CH;
  const __bf16* brow = vb + ((size_t)b * NSP + (size_t)e0 * 128) * CH;
  v8f acc = (v8f)0.f;
  for (int k0 = 0; k0 < NSP; k0 += 32) {
    __builtin_prefetch(arow + k0 + 512, 0, 1);
    __builtin_prefetch(brow + k0 + 512, 0, 1);
    v16bf a  = frag_a(arow, k0);
    v16bf bm = frag_b(brow, k0);
    acc = wmma_bf16(a, bm, acc);
  }
#endif

#pragma unroll
  for (int v = 0; v < 8; ++v) {
    int m, n; d_coords(v, m, n);
    int d = mt * 16 + m, e = nt * 16 + n;
    ctxT[(((size_t)b * 4 + h) * 32 + e) * 32 + d] = acc[v];
  }
}

// ===========================================================================
// K3: q softmax over d (axis=-2), stride 128*128 elems, in place, * inv_scale.
// grid: 2048 x 256; one thread per (b,h,s_local,c) group.
// ===========================================================================
__global__ void __launch_bounds__(256)
k_qsoft(__bf16* __restrict__ qb) {
  const int gid = blockIdx.x * 256 + threadIdx.x;   // < 8*4*128*128
  const int b  = gid >> 16;
  const int h  = (gid >> 14) & 3;
  const int sl = (gid >> 7) & 127;
  const int c  = gid & 127;
  __bf16* p = qb + ((size_t)b * NSP + (size_t)h * 4096 + sl) * CH + c;
  const size_t stride = (size_t)128 * CH;           // one d step

  float mx = -INFINITY;
#pragma unroll 4
  for (int d = 0; d < 32; ++d) mx = fmaxf(mx, (float)p[d * stride]);
  float sum = 0.f;
#pragma unroll 4
  for (int d = 0; d < 32; ++d) sum += expf((float)p[d * stride] - mx);
  const float rs = INV_SCALE / sum;
#pragma unroll 4
  for (int d = 0; d < 32; ++d) {
    float v = expf((float)p[d * stride] - mx) * rs;
    p[d * stride] = (__bf16)v;
  }
}

// ===========================================================================
// K4: attn = ctxT @ Q, out = attn @ w_out + b, LayerNorm, + x.
// grid: B*4*128 = 4096 WGs (b,h,s_local) x 256 threads (8 waves).
// ===========================================================================
__global__ void __launch_bounds__(256)
k_out(const __bf16* __restrict__ qb, const float* __restrict__ ctxT,
      const float* __restrict__ w_out, const float* __restrict__ b_out,
      const float* __restrict__ gamma, const float* __restrict__ beta,
      const float* __restrict__ x, float* __restrict__ out) {
  __shared__ __bf16 wto[128][CH];    // wto[c'][c] transposed w_out (32KB)
  __shared__ __bf16 ct[32][32];      // ctxT[e][d] (2KB)
  __shared__ __bf16 qt[CH][32];      // qt[c][d]   (8KB)
  __shared__ __bf16 at[32][CH];      // attn[e][c] (8KB)
  __shared__ float  yb[32][CH];      // pre-LN out (16KB)
  __shared__ float  rsum[32][8], rsq[32][8];

  const int tid  = threadIdx.x;
  const int lane = tid & 31;
  const int wave = tid >> 5;
  const int b  = blockIdx.x >> 9;
  const int h  = (blockIdx.x >> 7) & 3;
  const int sl = blockIdx.x & 127;

  // stage transposed w_out
#pragma unroll 4
  for (int t = 0; t < 64; ++t) {
    int idx = tid + 256 * t;                 // idx = c*128 + c' (coalesced)
    int c = idx >> 7, cp = idx & 127;
    wto[cp][c] = (__bf16)w_out[(size_t)c * CH + cp];
  }
  // stage ctxT[e][d]
#pragma unroll
  for (int t = 0; t < 4; ++t) {
    int idx = tid + 256 * t;                 // < 1024
    int e = idx >> 5, d = idx & 31;
    ct[e][d] = (__bf16)ctxT[(((size_t)b * 4 + h) * 32 + e) * 32 + d];
  }
  // stage Q block: qt[c][d] = qs[b, h*4096 + d*128 + sl, c]
#pragma unroll 4
  for (int t = 0; t < 16; ++t) {
    int idx = tid + 256 * t;                 // idx = d*128 + c
    int d = idx >> 7, c = idx & 127;
    qt[c][d] = qb[((size_t)b * NSP + (size_t)h * 4096 + (size_t)d * 128 + sl) * CH + c];
  }
  __syncthreads();

  const int mt    = wave >> 2;               // e half
  const int ntb   = (wave & 3) * 2;          // two n tiles per wave

  // GEMM1: attn[e][c] = ctxT[e][:] . qt-column c   (K = 32, single step)
  {
    const __bf16* arow = &ct[mt * 16 + (lane & 15)][0];
    v16bf a = frag_a(arow, 0);
#pragma unroll
    for (int j = 0; j < 2; ++j) {
      int nt = ntb + j;
      v16bf bf = frag_b(&qt[nt * 16 + (lane & 15)][0], 0);
      v8f acc = wmma_bf16(a, bf, (v8f)0.f);
#pragma unroll
      for (int v = 0; v < 8; ++v) {
        int m, n; d_coords(v, m, n);
        at[mt * 16 + m][nt * 16 + n] = (__bf16)acc[v];
      }
    }
  }
  __syncthreads();

  // GEMM2: y[e][c'] = attn[e][:] @ w_out[:,c']   (K = 128)
  {
    v8f acc2[2] = {(v8f)0.f, (v8f)0.f};
    const __bf16* arow = &at[mt * 16 + (lane & 15)][0];
#pragma unroll
    for (int k0 = 0; k0 < 128; k0 += 32) {
      v16bf a = frag_a(arow, k0);
#pragma unroll
      for (int j = 0; j < 2; ++j) {
        v16bf bf = frag_b(&wto[(ntb + j) * 16 + (lane & 15)][0], k0);
        acc2[j] = wmma_bf16(a, bf, acc2[j]);
      }
    }
#pragma unroll
    for (int j = 0; j < 2; ++j)
#pragma unroll
      for (int v = 0; v < 8; ++v) {
        int m, n; d_coords(v, m, n);
        yb[mt * 16 + m][(ntb + j) * 16 + n] = acc2[j][v];
      }
  }
  __syncthreads();

  // bias + LayerNorm(eps=1e-3) + residual. 8 threads per row e.
  const int e = tid >> 3;
  const int g = tid & 7;
  float s = 0.f, q = 0.f;
#pragma unroll
  for (int i = 0; i < 16; ++i) {
    int cp = g * 16 + i;
    float v = yb[e][cp] + b_out[cp];
    yb[e][cp] = v;
    s += v; q += v * v;
  }
  rsum[e][g] = s; rsq[e][g] = q;
  __syncthreads();

  float ts = 0.f, tq = 0.f;
#pragma unroll
  for (int i = 0; i < 8; ++i) { ts += rsum[e][i]; tq += rsq[e][i]; }
  const float mu   = ts * (1.f / 128.f);
  const float var  = tq * (1.f / 128.f) - mu * mu;
  const float rstd = rsqrtf(var + LN_EPS);
  const size_t srow = ((size_t)b * NSP + (size_t)(h * 32 + e) * 128 + sl) * CH;
#pragma unroll
  for (int i = 0; i < 16; ++i) {
    int cp = g * 16 + i;
    float o = (yb[e][cp] - mu) * rstd * gamma[cp] + beta[cp] + x[srow + cp];
    out[srow + cp] = o;
  }
}

// ===========================================================================
extern "C" void kernel_launch(void* const* d_in, const int* in_sizes, int n_in,
                              void* d_out, int out_size, void* d_ws, size_t ws_size,
                              hipStream_t stream) {
  (void)in_sizes; (void)n_in; (void)out_size; (void)ws_size;
  const float* x     = (const float*)d_in[0];
  const float* w_qkv = (const float*)d_in[1];
  const float* w_out = (const float*)d_in[2];
  const float* b_out = (const float*)d_in[3];
  const float* gamma = (const float*)d_in[4];
  const float* beta  = (const float*)d_in[5];
  float* out = (float*)d_out;

  // workspace carve (bf16 intermediates): 3 * 32MB + ctx
  char* w = (char*)d_ws;
  const size_t plane = (size_t)8 * NSP * CH * sizeof(__bf16);  // 32 MB
  __bf16* qb = (__bf16*)w;            w += plane;
  __bf16* ks = (__bf16*)w;            w += plane;
  __bf16* vb = (__bf16*)w;            w += plane;
  float*  ctxT = (float*)w;           // 8*4*32*32 f32

  k_qkv  <<<dim3(8 * 128), dim3(256), 0, stream>>>(x, w_qkv, qb, ks, vb);
  k_ctx  <<<dim3(32),      dim3(128), 0, stream>>>(ks, vb, ctxT);
  k_qsoft<<<dim3(2048),    dim3(256), 0, stream>>>(qb);
  k_out  <<<dim3(4096),    dim3(256), 0, stream>>>(qb, ctxT, w_out, b_out,
                                                   gamma, beta, x, out);
}
